// CharRNN_86792699118070
// MI455X (gfx1250) — matrix operand order repaired and checked by reading, not compiled, not run on hardware
//
#include <hip/hip_runtime.h>
#include <hip/hip_bf16.h>

// ================= feature detection (compile-safe guards) =================
#ifndef __has_builtin
#define __has_builtin(x) 0
#endif

#if __has_builtin(__builtin_amdgcn_global_load_async_to_lds_b128)
#define HAS_ASYNC_LDS 1
#else
#define HAS_ASYNC_LDS 0
#endif

#if __has_builtin(__builtin_amdgcn_tensor_load_to_lds)
#define HAS_TDM 1
#else
#define HAS_TDM 0
#endif

#if __has_include(<hip/amd_detail/amd_gfx1250_TDM.h>)
#define TDM_SIX_ARGS 1   // therock/clang-23 toolchain: 6-arg builtin
#else
#define TDM_SIX_ARGS 0   // ROCm 7.2 / clang-22: 5-arg builtin
#endif

#define __AS1 __attribute__((address_space(1)))
#define __AS3 __attribute__((address_space(3)))

// ================= types =================
typedef __attribute__((ext_vector_type(16))) __bf16 v16bf;
typedef __attribute__((ext_vector_type(8)))  __bf16 v8bf;
typedef __attribute__((ext_vector_type(8)))  float  v8f;
typedef __attribute__((ext_vector_type(4)))  unsigned int u32x4;
typedef __attribute__((ext_vector_type(4)))  int i32x4;
typedef __attribute__((ext_vector_type(8)))  int i32x8;

__device__ __forceinline__ __bf16 f2bf(float f) {
    unsigned u = __builtin_bit_cast(unsigned, f);
    unsigned r = u + 0x7FFFu + ((u >> 16) & 1u);   // round-to-nearest-even
    return __builtin_bit_cast(__bf16, (unsigned short)(r >> 16));
}

__device__ __forceinline__ v16bf cat8(v8bf lo, v8bf hi) {
    return __builtin_shufflevector(lo, hi, 0,1,2,3,4,5,6,7,8,9,10,11,12,13,14,15);
}

__device__ __forceinline__ v8f wmma_bf16(v16bf a, v16bf b, v8f c) {
    return __builtin_amdgcn_wmma_f32_16x16x32_bf16(false, a, false, b, (short)0, c, false, false);
}

// 16-byte global -> LDS copy: async DMA path on CDNA5, direct copy fallback
__device__ __forceinline__ void copy16(const __bf16* g, __bf16* l) {
#if HAS_ASYNC_LDS
    __builtin_amdgcn_global_load_async_to_lds_b128(
        (__AS1 i32x4*)(void*)const_cast<__bf16*>(g), (__AS3 i32x4*)l, 0, 0);
#else
    *(v8bf*)l = *(const v8bf*)g;
#endif
}

template <int N> __device__ __forceinline__ void wait_async() {
#if HAS_ASYNC_LDS
 #if __has_builtin(__builtin_amdgcn_s_wait_asynccnt)
    __builtin_amdgcn_s_wait_asynccnt(N);
 #else
    asm volatile("s_wait_asynccnt %0" :: "n"(N) : "memory");
 #endif
#endif
}

__device__ __forceinline__ void wait_tensorcnt0() {
#if __has_builtin(__builtin_amdgcn_s_wait_tensorcnt)
    __builtin_amdgcn_s_wait_tensorcnt(0);
#else
    asm volatile("s_wait_tensorcnt 0x0" ::: "memory");
#endif
}

// ================= kernel: transpose fp32 weight -> bf16 (Wt[c*rows+r] = W[r*cols+c]) ===========
__global__ void k_transpose_bf16(const float* __restrict__ W, __bf16* __restrict__ Wt,
                                 int rows, int cols) {
    int idx = blockIdx.x * blockDim.x + threadIdx.x;
    int total = rows * cols;
    int stride = gridDim.x * blockDim.x;
    for (; idx < total; idx += stride) {
        int c = idx / rows;
        int r = idx - c * rows;
        Wt[idx] = f2bf(W[(size_t)r * cols + c]);
    }
}

// ================= kernel: fused embedding gather + fp32->bf16 =================
__global__ void k_embed(const int* __restrict__ x, const float* __restrict__ emb,
                        __bf16* __restrict__ Xe) {
    int idx = blockIdx.x * blockDim.x + threadIdx.x;
    int stride = gridDim.x * blockDim.x;
    const int total = 16384 * 1024;
    for (; idx < total; idx += stride) {
        int row = idx >> 10;
        int col = idx & 1023;
        Xe[idx] = f2bf(emb[(size_t)x[row] * 1024 + col]);
    }
}

// ================= block-tiled bf16 GEMM: C(MxN) = A(MxK) * B(KxN), fp32 out ===========
// 256 threads = 8 waves; block tile 128x128, K-chunk 32, async double-buffered LDS staging.
// Wave grid 2(M)x4(N): each wave computes 64x32 = 4x2 WMMA tiles.
#define LDA 40    // A LDS row stride in halves (80B: 16B-aligned, conflict-free over 16 rows)
#define LDB 136   // B LDS row stride in halves (272B: 16B-aligned, 2-way max)
__global__ void __launch_bounds__(256)
k_gemm_bf16(const __bf16* __restrict__ A, const __bf16* __restrict__ B,
            float* __restrict__ C, int M, int N, int K) {
    __shared__ __bf16 As[2][128 * LDA];
    __shared__ __bf16 Bs[2][32 * LDB];
    const int tid = threadIdx.x, wave = tid >> 5, lane = tid & 31;
    const int hsel = lane >> 4, ml = lane & 15;
    const int nb = N >> 7;
    const int m0 = (blockIdx.x / nb) << 7;
    const int n0 = (blockIdx.x % nb) << 7;
    const int wm = wave & 1, wn = wave >> 1;

    auto issue = [&](int s, int kc) {
        #pragma unroll
        for (int i = 0; i < 2; ++i) {            // A tile: 128 rows x 4 16B-chunks
            int j = tid + 256 * i;
            int row = j >> 2, c = (j & 3) * 8;
            copy16(A + (size_t)(m0 + row) * K + kc + c, &As[s][row * LDA + c]);
        }
        #pragma unroll
        for (int i = 0; i < 2; ++i) {            // B tile: 32 rows x 16 16B-chunks
            int j = tid + 256 * i;
            int row = j >> 4, c = (j & 15) * 8;
            copy16(B + (size_t)(kc + row) * N + n0 + c, &Bs[s][row * LDB + c]);
        }
    };

    v8f acc[4][2] = {};
    issue(0, 0);
    const int nk = K >> 5;
    for (int i = 0; i < nk; ++i) {
        if (i + 1 < nk) { issue((i + 1) & 1, (i + 1) << 5); wait_async<4>(); }
        else            { wait_async<0>(); }
        __syncthreads();
        const int s = i & 1;
        v16bf af[4], bg[2];
        #pragma unroll
        for (int st = 0; st < 4; ++st) {
            const __bf16* p = &As[s][(wm * 64 + st * 16 + ml) * LDA + 8 * hsel];
            af[st] = cat8(*(const v8bf*)p, *(const v8bf*)(p + 16));
        }
        #pragma unroll
        for (int j = 0; j < 2; ++j) {
            const __bf16* p = &Bs[s][8 * hsel * LDB + wn * 32 + j * 16 + ml];
            #pragma unroll
            for (int e = 0; e < 8; ++e) { bg[j][e] = p[e * LDB]; bg[j][8 + e] = p[(16 + e) * LDB]; }
        }
        #pragma unroll
        for (int st = 0; st < 4; ++st)
            #pragma unroll
            for (int j = 0; j < 2; ++j)
                acc[st][j] = wmma_bf16(af[st], bg[j], acc[st][j]);
        __syncthreads();
    }
    #pragma unroll
    for (int st = 0; st < 4; ++st) {
        int rb = m0 + wm * 64 + st * 16 + 8 * hsel;
        #pragma unroll
        for (int j = 0; j < 2; ++j) {
            int col = n0 + wn * 32 + j * 16 + ml;
            #pragma unroll
            for (int r = 0; r < 8; ++r)
                C[(size_t)(rb + r) * N + col] = acc[st][j][r];
        }
    }
}

// ================= RNN recurrence: 16 blocks, each owns 64 N-columns ===========
#define RNN_BLOCKS 16
#define LDBR 72   // W_h LDS slice row stride in halves (64 + 8 pad)

__device__ __forceinline__ void grid_barrier(unsigned* cnt, unsigned* gen) {
    __threadfence();
    __syncthreads();
    if (threadIdx.x == 0) {
        unsigned g = __hip_atomic_load(gen, __ATOMIC_ACQUIRE, __HIP_MEMORY_SCOPE_AGENT);
        unsigned a = __hip_atomic_fetch_add(cnt, 1u, __ATOMIC_ACQ_REL, __HIP_MEMORY_SCOPE_AGENT);
        if (a == RNN_BLOCKS - 1) {
            __hip_atomic_store(cnt, 0u, __ATOMIC_RELAXED, __HIP_MEMORY_SCOPE_AGENT);
            __hip_atomic_fetch_add(gen, 1u, __ATOMIC_ACQ_REL, __HIP_MEMORY_SCOPE_AGENT);
        } else {
            while (__hip_atomic_load(gen, __ATOMIC_ACQUIRE, __HIP_MEMORY_SCOPE_AGENT) == g)
                __builtin_amdgcn_s_sleep(1);
        }
    }
    __syncthreads();
}

__global__ void k_rnn_init(const float* __restrict__ h0, __bf16* __restrict__ hb0,
                           unsigned* __restrict__ bar) {
    int idx = blockIdx.x * blockDim.x + threadIdx.x;
    int stride = gridDim.x * blockDim.x;
    for (int i = idx; i < 32 * 1024; i += stride) hb0[i] = f2bf(h0[i]);
    if (idx < 2) bar[idx] = 0u;
}

// h_t = tanh(pre_t + h_{t-1} @ W_h^T). W_h^T column slice lives in LDS for all 512 steps
// (loaded once via the Tensor Data Mover when available); h is exchanged through global
// memory with a 16-block grid barrier per step.
__global__ void __launch_bounds__(256)
k_rnn(const float* __restrict__ pre, const __bf16* __restrict__ WhT,
      __bf16* __restrict__ hb0, __bf16* __restrict__ hb1,
      __bf16* __restrict__ outs, float* __restrict__ hfinal,
      unsigned* __restrict__ bar) {
    extern __shared__ __bf16 rnn_lds[];
    __bf16* Bsl = rnn_lds;                 // [1024][LDBR] W_h^T slice (cols ncol0..+64)
    __bf16* Hs  = rnn_lds + 1024 * LDBR;   // [32][1024] h_{t-1}, XOR-swizzled
    const int tid = threadIdx.x, wave = tid >> 5, lane = tid & 31;
    const int hsel = lane >> 4, ml = lane & 15;
    const int ncol0 = blockIdx.x * 64;
    const int mt = wave & 1, ntl = wave >> 1;     // 2 M-tiles x 4 N-tiles = 8 waves

    // ---- one-time preload of this block's 1024x64 W_h^T slice into LDS ----
#if HAS_TDM
    if (tid == 0) {
        unsigned lds_byte = (unsigned)(size_t)(__AS3 void*)Bsl;
        unsigned long long ga = (unsigned long long)(const void*)(WhT + ncol0);
        u32x4 g0 = { 1u,                                  // count=1
                     lds_byte,                            // lds_addr
                     (unsigned)ga,                        // global_addr[31:0]
                     (unsigned)((ga >> 32) & 0x1FFFFFFu) | (2u << 30) }; // addr[56:32] | type=2
        i32x8 g1 = { (1 << 16) | (1 << 20) | (4 << 22) | (3 << 25),
                     //  data_size=2B | pad_enable | pad_interval=32dw | pad_amount=4dw
                     (int)(1024u << 16),                  // tensor_dim0 = 1024 (low16 in [31:16])
                     (int)(1024u << 16),                  // dim0 hi=0 | tensor_dim1 = 1024
                     (int)(64u << 16),                    // dim1 hi=0 | tile_dim0 = 64
                     1024,                                // tile_dim1 = 1024 | tile_dim2 = 0
                     1024,                                // tensor_dim0_stride = 1024
                     0, 0 };
        i32x4 z4 = { 0, 0, 0, 0 };
#if TDM_SIX_ARGS
        i32x8 z8 = { 0, 0, 0, 0, 0, 0, 0, 0 };
        __builtin_amdgcn_tensor_load_to_lds(g0, g1, z4, z4, z8, 0);
#else
        __builtin_amdgcn_tensor_load_to_lds(g0, g1, z4, z4, 0);
#endif
        wait_tensorcnt0();
    }
#else
    for (int j = tid; j < 1024 * 8; j += 256) {   // 8192 16B-chunks
        int k = j >> 3, c = (j & 7) * 8;
        copy16(WhT + (size_t)k * 1024 + ncol0 + c, Bsl + k * LDBR + c);
    }
    wait_async<0>();
#endif
    __syncthreads();

    for (int t = 0; t < 512; ++t) {
        const __bf16* hprev = (t & 1) ? hb1 : hb0;
        __bf16*       hnext = (t & 1) ? hb0 : hb1;
        // stage h_{t-1} (64KB) into LDS with XOR swizzle, via async-to-LDS
        for (int j = tid; j < 32 * 128; j += 256) {
            int row = j >> 7, c8 = (j & 127) * 8;
            copy16(hprev + row * 1024 + c8, Hs + row * 1024 + (c8 ^ ((row & 15) << 3)));
        }
        wait_async<0>();
        __syncthreads();

        v8f acc = {};
        for (int kc = 0; kc < 1024; kc += 32) {
            int row = mt * 16 + ml, sw = ml << 3;
            const __bf16* ab = Hs + row * 1024;
            v16bf a = cat8(*(const v8bf*)(ab + ((kc +      8 * hsel) ^ sw)),
                           *(const v8bf*)(ab + ((kc + 16 + 8 * hsel) ^ sw)));
            const __bf16* bb = Bsl + (kc + 8 * hsel) * LDBR + ntl * 16 + ml;
            v16bf b;
            #pragma unroll
            for (int e = 0; e < 8; ++e) { b[e] = bb[e * LDBR]; b[8 + e] = bb[(16 + e) * LDBR]; }
            acc = wmma_bf16(a, b, acc);
        }

        int n = ncol0 + ntl * 16 + ml;
        #pragma unroll
        for (int r = 0; r < 8; ++r) {
            int bi = mt * 16 + r + 8 * hsel;                 // batch index
            size_t rowoff = ((size_t)bi * 512 + t) * 1024;   // [b, t, :]
            float v = tanhf(acc[r] + pre[rowoff + n]);
            __bf16 vb = f2bf(v);
            outs[rowoff + n] = vb;
            hnext[bi * 1024 + n] = vb;
            if (t == 511) hfinal[(size_t)bi * 1024 + n] = v;
        }
        __syncthreads();                      // all Hs reads done before next staging
        grid_barrier(bar, bar + 1);           // h_t globally visible
    }
}

// ================= launch =================
extern "C" void kernel_launch(void* const* d_in, const int* in_sizes, int n_in,
                              void* d_out, int out_size, void* d_ws, size_t ws_size,
                              hipStream_t stream) {
    (void)in_sizes; (void)n_in; (void)out_size; (void)ws_size;
    const int*   x    = (const int*)d_in[0];
    const float* hid0 = (const float*)d_in[1];
    const float* emb  = (const float*)d_in[2];
    const float* W_h  = (const float*)d_in[3];
    const float* W_e  = (const float*)d_in[4];
    const float* W_o  = (const float*)d_in[5];

    float* logits = (float*)d_out;                       // [32*512, 8192]
    float* hfinal = logits + (size_t)32 * 512 * 8192;    // [32, 1024]

    char* ws = (char*)d_ws;
    float*  pre = (float*)ws;   ws += (size_t)16384 * 1024 * sizeof(float);   // 64 MB
    __bf16* Xe  = (__bf16*)ws;  ws += (size_t)16384 * 1024 * sizeof(__bf16);  // 32 MB (reused as outputs)
    __bf16* WeT = (__bf16*)ws;  ws += (size_t)1024 * 1024 * sizeof(__bf16);   // 2 MB
    __bf16* WhT = (__bf16*)ws;  ws += (size_t)1024 * 1024 * sizeof(__bf16);   // 2 MB
    __bf16* WoT = (__bf16*)ws;  ws += (size_t)1024 * 8192 * sizeof(__bf16);   // 16 MB
    __bf16* hb0 = (__bf16*)ws;  ws += (size_t)32 * 1024 * sizeof(__bf16);
    __bf16* hb1 = (__bf16*)ws;  ws += (size_t)32 * 1024 * sizeof(__bf16);
    unsigned* bar = (unsigned*)ws;

    const size_t rnn_lds_bytes = (size_t)(1024 * LDBR + 32 * 1024) * sizeof(__bf16); // 208 KB
    (void)hipFuncSetAttribute((const void*)k_rnn,
                              hipFuncAttributeMaxDynamicSharedMemorySize,
                              (int)rnn_lds_bytes);

    // weights -> bf16 transposed (B layout)
    k_transpose_bf16<<<2048, 256, 0, stream>>>(W_e, WeT, 1024, 1024);
    k_transpose_bf16<<<2048, 256, 0, stream>>>(W_h, WhT, 1024, 1024);
    k_transpose_bf16<<<8192, 256, 0, stream>>>(W_o, WoT, 8192, 1024);

    // embedding gather + convert
    k_embed<<<8192, 256, 0, stream>>>(x, emb, Xe);

    // pre = Xe @ W_e^T : M=16384 N=1024 K=1024; 128x128 tiles -> 128*8 blocks
    k_gemm_bf16<<<128 * 8, 256, 0, stream>>>(Xe, WeT, pre, 16384, 1024, 1024);

    // recurrence: 16 blocks x 64 columns; outputs overwrite Xe
    k_rnn_init<<<64, 256, 0, stream>>>(hid0, hb0, bar);
    k_rnn<<<RNN_BLOCKS, 256, rnn_lds_bytes, stream>>>(pre, WhT, hb0, hb1, Xe, hfinal, bar);

    // logits = outputs @ W_o^T : M=16384 N=8192 K=1024; 128*64 blocks
    k_gemm_bf16<<<128 * 64, 256, 0, stream>>>(Xe, WoT, logits, 16384, 8192, 1024);
}
